// SingleLayerConv_67654324846783
// MI455X (gfx1250) — matrix operand order repaired and checked by reading, not compiled, not run on hardware
//
#include <hip/hip_runtime.h>
#include <hip/hip_bf16.h>

// ---------------------------------------------------------------------------
// Sparse->dense 3D conv (k=3, VALID) as implicit GEMM on CDNA5:
//   TDM (tensor_load_to_lds) -> LDS slab -> v_wmma_f32_16x16x32_bf16.
//
// Roofline: grid bf16 = 99 MB (L2-resident). Direct-from-L2 B operands cost
// ~2.2 GB of L2 reads; staging an (11z x 3y x 18x x 32ch) slab per workgroup
// via TDM cuts global/L2 reads to ~330 MB (~HBM floor) and serves the 27x
// reuse from LDS. TDM padding (4 dw per 16 dw -> 80 B x-row stride) makes the
// ds_load_b128 pattern provably 64-bank conflict-free.
// ---------------------------------------------------------------------------

#define DI   11
#define HI   400
#define WI   352
#define CIN  32
#define COUT 16
#define DO_  9
#define HO   398
#define WO   350
#define NTAP 27

#define XT     16                 // x-outputs per workgroup
#define XSPAN  18                 // input x span (XT + 2)
#define TILES_X 22                // ceil(350/16)
#define ROWB   80                 // padded LDS bytes per x position (64 + 16 pad)
#define LROW   (XSPAN * ROWB)     // LDS bytes per (z,y) line = 1440
#define SLAB_B (DI * 3 * LROW)    // 47,520 B

typedef __attribute__((ext_vector_type(16))) __bf16 v16bf;
typedef __attribute__((ext_vector_type(8)))  float  v8f;
typedef __attribute__((ext_vector_type(4)))  unsigned int v4u;
typedef __attribute__((ext_vector_type(8)))  int          v8i;
typedef __attribute__((ext_vector_type(4)))  int          v4i;

static constexpr long long GRID_ELEMS = (long long)DI * HI * WI * CIN;   // bf16 elems
static constexpr long long GRID_BYTES = GRID_ELEMS * 2;                  // 99,123,200
static constexpr long long WA_OFF     = (GRID_BYTES + 255) & ~255ll;     // swizzled weights

__device__ __forceinline__ v16bf load_v16bf(const __bf16* p) {
    return *reinterpret_cast<const v16bf*>(p);  // 32B aligned -> 2x global_load_b128
}

// --- 1) zero the dense bf16 grid (16B stores) ------------------------------
__global__ void zero_grid_kernel(uint4* __restrict__ g, long long n16) {
    long long i = (long long)blockIdx.x * blockDim.x + threadIdx.x;
    if (i < n16) g[i] = make_uint4(0u, 0u, 0u, 0u);
}

// --- 2) scatter voxel features into the grid, f32 -> bf16 ------------------
__global__ void scatter_kernel(const float* __restrict__ vf,
                               const int* __restrict__ coors,
                               __bf16* __restrict__ grid, int nv) {
    int t = blockIdx.x * blockDim.x + threadIdx.x;
    int v = t >> 2, q = t & 3;
    if (v >= nv) return;
    int z = coors[v * 4 + 1];
    int y = coors[v * 4 + 2];
    int x = coors[v * 4 + 3];
    const float* src = vf + (long long)v * CIN + q * 8;
    __bf16* dst = grid + (((long long)z * HI + y) * WI + x) * CIN + q * 8;
    union { __bf16 h[8]; uint4 u4; } pk;
#pragma unroll
    for (int i = 0; i < 8; ++i) pk.h[i] = (__bf16)src[i];
    *reinterpret_cast<uint4*>(dst) = pk.u4;
}

// --- 3) pre-swizzle weights into the ISA A-operand layout ------------------
// A = per-tap cout(16) x cin(32). 16-bit 16x32 A layout:
//   lanes 0-15 : M=lane,    K = {0..7, 16..23}
//   lanes 16-31: M=lane-16, K = {8..15, 24..31}
__global__ void swizzle_w_kernel(const float* __restrict__ w,
                                 __bf16* __restrict__ wA) {
    int t = blockIdx.x * blockDim.x + threadIdx.x;  // tap*32 + lane
    if (t >= NTAP * 32) return;
    int tap   = t >> 5;
    int lane  = t & 31;
    int m     = lane & 15;
    int kbase = (lane >> 4) * 8;
    __bf16* dst = wA + (long long)t * 16;
#pragma unroll
    for (int i = 0; i < 16; ++i) {
        int k = kbase + (i & 7) + ((i >> 3) * 16);
        dst[i] = (__bf16)w[((long long)tap * CIN + k) * COUT + m];
    }
}

// --- 4) conv: TDM slab -> LDS, WMMA from LDS -------------------------------
// Workgroup = 3 waves (96 thr), covers one yo and 16 x-outputs, ALL 9 zo.
// Wave w owns zo in {3w, 3w+1, 3w+2}; A is loaded once per tap, reused 3x.
__global__ void __launch_bounds__(96)
conv_wmma_kernel(const __bf16* __restrict__ grid,
                 const __bf16* __restrict__ wA,
                 float* __restrict__ out) {
    __shared__ __align__(16) unsigned char slab[SLAB_B];

    int wg = blockIdx.x;
    int xt = wg % TILES_X;
    int yo = wg / TILES_X;
    int xbase = xt * XT;

    // ---- TDM: load (11 z) x (3 y) x (XSPAN x * 32 ch) bf16 slab into LDS.
    // D#: data_size=4B. dim0 = fused (x,ch) = XSPAN*16 dwords, dim1 = y
    // (stride 5632 dw), dim2 = z (stride 400*5632 dw). pad: 4 dw per 16 dw.
    if ((threadIdx.x >> 5) == 0) {
        unsigned lds_base = (unsigned)(size_t)(void*)slab;
        unsigned long long ga = (unsigned long long)(size_t)grid
                              + ((unsigned long long)yo * WI + (unsigned)xbase) * (CIN * 2);
        v4u g0;
        g0.x = 1u;                                   // count=1 (valid user D#)
        g0.y = lds_base;                             // lds_addr
        g0.z = (unsigned)ga;                         // global_addr[31:0]
        g0.w = (unsigned)((ga >> 32) & 0x1FFFFFFu)   // global_addr[56:32]
             | (2u << 30);                           // type=2 (image)
        v8i g1;
        g1[0] = (int)((2u << 16)     // data_size = 4B
                    | (1u << 20)     // pad_enable
                    | (3u << 22)     // pad_interval: 16 dwords
                    | (3u << 25));   // pad_amount:   4 dwords
        g1[1] = (int)((WI * 16) << 16);              // tensor_dim0[15:0] = 5632 dw
        g1[2] = (int)(HI << 16);                     // tensor_dim0[31:16]=0 | tensor_dim1=400
        g1[3] = (int)((XSPAN * 16) << 16);           // tensor_dim1 hi=0 | tile_dim0=288 dw
        g1[4] = (int)(3u | ((unsigned)DI << 16));    // tile_dim1=3 (y) | tile_dim2=11 (z)
        g1[5] = WI * 16;                             // tensor_dim0_stride = 5632 dw
        g1[6] = (int)(((unsigned)(HI * WI * 16) & 0xFFFFu) << 16); // stride0 hi | stride1 lo
        g1[7] = (int)((unsigned)(HI * WI * 16) >> 16);             // stride1 hi
        v4i g2; g2[0] = DI; g2[1] = 0; g2[2] = 0; g2[3] = 0;       // tensor_dim2=11
        v4i g3; g3[0] = 0;  g3[1] = 0; g3[2] = 0; g3[3] = 0;
#if defined(__clang_major__) && (__clang_major__ >= 23)
        v8i gz; gz[0]=0; gz[1]=0; gz[2]=0; gz[3]=0; gz[4]=0; gz[5]=0; gz[6]=0; gz[7]=0;
        __builtin_amdgcn_tensor_load_to_lds(g0, g1, g2, g3, gz, 0);
#else
        __builtin_amdgcn_tensor_load_to_lds(g0, g1, g2, g3, 0);
#endif
        __builtin_amdgcn_s_wait_tensorcnt(0);
    }
    __syncthreads();

    int wave = threadIdx.x >> 5;   // 0..2 -> zo group
    int lane = threadIdx.x & 31;
    int col  = lane & 15;          // B/D column = x within tile
    int half = lane >> 4;          // K/M half

    v8f c[3];
#pragma unroll
    for (int i = 0; i < 3; ++i) c[i] = (v8f){};

    for (int tap = 0; tap < NTAP; ++tap) {
        int dz = tap / 9, dy = (tap / 3) % 3, dx = tap % 3;
        // A: weights, pre-swizzled, L0-resident; loaded once, reused 3x.
        v16bf a = load_v16bf(wA + ((long long)tap * 32 + lane) * 16);
        int xi = col + dx;                       // 0..17 within slab
        unsigned off = (unsigned)((dz * 3 + dy) * LROW + xi * ROWB + half * 16);
#pragma unroll
        for (int i = 0; i < 3; ++i) {
            int zo = wave * 3 + i;
            const unsigned char* p = slab + (unsigned)(zo * 3 * LROW) + off;
            union { uint4 q[2]; v16bf v; } b;
            b.q[0] = *reinterpret_cast<const uint4*>(p);        // ds_load_b128
            b.q[1] = *reinterpret_cast<const uint4*>(p + 32);   // ds_load_b128
            c[i] = __builtin_amdgcn_wmma_f32_16x16x32_bf16(
                       false, a, false, b.v, (short)0, c[i], false, false);
        }
    }

    // D: lane holds column N=col (x), rows M = r + 8*half (cout). Coalesced.
    int xo = xbase + col;
    if (xo < WO) {
#pragma unroll
        for (int i = 0; i < 3; ++i) {
            int zo = wave * 3 + i;
#pragma unroll
            for (int r = 0; r < 8; ++r) {
                int co = r + half * 8;
                out[(((long long)co * DO_ + zo) * HO + yo) * WO + xo] = c[i][r];
            }
        }
    }
}

// ---------------------------------------------------------------------------
extern "C" void kernel_launch(void* const* d_in, const int* in_sizes, int n_in,
                              void* d_out, int out_size, void* d_ws, size_t ws_size,
                              hipStream_t stream) {
    const float* vf    = (const float*)d_in[0];  // (60000, 32) f32
    const int*   coors = (const int*)d_in[1];    // (60000, 4) i32: b,z,y,x
    const float* w     = (const float*)d_in[2];  // (3,3,3,32,16) f32

    float*  out  = (float*)d_out;
    char*   ws   = (char*)d_ws;
    __bf16* grid = (__bf16*)ws;                  // 99.1 MB bf16 dense grid
    __bf16* wA   = (__bf16*)(ws + WA_OFF);       // 27 KB swizzled weights

    long long n16 = GRID_BYTES / 16;
    zero_grid_kernel<<<(unsigned)((n16 + 255) / 256), 256, 0, stream>>>((uint4*)ws, n16);

    int nv = in_sizes[0] / CIN;  // 60000
    scatter_kernel<<<(nv * 4 + 255) / 256, 256, 0, stream>>>(vf, coors, grid, nv);

    swizzle_w_kernel<<<(NTAP * 32 + 255) / 256, 256, 0, stream>>>(w, wA);

    // 398 yo x 22 x-tiles workgroups, 96 threads (3 waves) each.
    conv_wmma_kernel<<<HO * TILES_X, 96, 0, stream>>>(grid, wA, out);
}
